// NERModel_75935021793906
// MI455X (gfx1250) — matrix operand order repaired and checked by reading, not compiled, not run on hardware
//
#include <hip/hip_runtime.h>
#include <hip/hip_bf16.h>

// Problem constants (B, S, H, L) from the reference.
#define B_   16
#define S_   512
#define H_   768
#define G4_  3072   // 4*H
#define H2_  1536   // 2*H
#define L_   25

typedef __bf16 bf16;
typedef __attribute__((ext_vector_type(16))) __bf16 v16bf;
typedef __attribute__((ext_vector_type(8)))  float  v8f;

union Frag16 { uint4 u[2]; v16bf v; };   // 32 bytes: one lane's A/B fragment

__device__ __forceinline__ float sigmoidf_(float x) { return 1.0f / (1.0f + __expf(-x)); }

// ---------------------------------------------------------------------------
// Grid-wide barrier across the workgroups of one LSTM direction.
// cnt/gen live in d_ws and are hipMemsetAsync'd to 0 before each launch.
// ---------------------------------------------------------------------------
__device__ __forceinline__ void grid_sync(int* cnt, int* gen, int nwg) {
  __syncthreads();
  if (threadIdx.x == 0) {
    __threadfence();
    int g = __hip_atomic_load(gen, __ATOMIC_RELAXED, __HIP_MEMORY_SCOPE_AGENT);
    int arrived = __hip_atomic_fetch_add(cnt, 1, __ATOMIC_ACQ_REL, __HIP_MEMORY_SCOPE_AGENT);
    if (arrived == nwg - 1) {
      __hip_atomic_store(cnt, 0, __ATOMIC_RELAXED, __HIP_MEMORY_SCOPE_AGENT);
      __hip_atomic_fetch_add(gen, 1, __ATOMIC_ACQ_REL, __HIP_MEMORY_SCOPE_AGENT);
    } else {
      while (__hip_atomic_load(gen, __ATOMIC_RELAXED, __HIP_MEMORY_SCOPE_AGENT) == g) {
        __builtin_amdgcn_s_sleep(2);
      }
    }
  }
  __syncthreads();
}

// ---------------------------------------------------------------------------
// f32 -> bf16 cast (grid-stride)
// ---------------------------------------------------------------------------
__global__ void __launch_bounds__(256)
cast_bf16_kernel(const float* __restrict__ src, bf16* __restrict__ dst, size_t n) {
  for (size_t i = (size_t)blockIdx.x * blockDim.x + threadIdx.x; i < n;
       i += (size_t)gridDim.x * blockDim.x)
    dst[i] = (bf16)src[i];
}

// ---------------------------------------------------------------------------
// XG = A(bf16, M x K) @ W(bf16, N x K)^T + (b0 + b1), C f32 (M x 3072).
// Block = 256 thr (8 waves). Wave computes 16 rows x 64 cols (4 WMMA tiles).
// grid = (N/512, M/16). Two-deep software pipeline: load fragment set Y for
// k+32 while issuing the 4 WMMAs on set X -> loads overlap the WMMA chain.
// ---------------------------------------------------------------------------
__global__ void __launch_bounds__(256)
gemm_xg_kernel(const bf16* __restrict__ A, const bf16* __restrict__ W,
               const float* __restrict__ b0, const float* __restrict__ b1,
               float* __restrict__ C, int K)
{
  const int wave = threadIdx.x >> 5;
  const int lane = threadIdx.x & 31;
  const int hi   = lane >> 4;       // half-wave select (K split: 16 contiguous K per half)
  const int lo   = lane & 15;
  const int m0   = blockIdx.y * 16;
  const int n0   = blockIdx.x * 512 + wave * 64;

  const v8f vzero = {0.f,0.f,0.f,0.f,0.f,0.f,0.f,0.f};
  v8f acc0 = vzero, acc1 = vzero, acc2 = vzero, acc3 = vzero;

  // A fragment: row = m0+lo; 16 contiguous K per lane: K {0..15} / {16..31}
  const bf16* pA  = A + (size_t)(m0 + lo) * K + hi * 16;
  // B fragment: col n; 16 contiguous K per lane (W is N x K row-major = B^T).
  // Scalar pointers (not an array!) so they stay in SGPR/VGPR and the loads
  // keep the global address space (global_load_b128, not flat_load).
  const bf16* pW0 = W + (size_t)(n0 +  0 + lo) * K + hi * 16;
  const bf16* pW1 = W + (size_t)(n0 + 16 + lo) * K + hi * 16;
  const bf16* pW2 = W + (size_t)(n0 + 32 + lo) * K + hi * 16;
  const bf16* pW3 = W + (size_t)(n0 + 48 + lo) * K + hi * 16;

  auto lda = [&](int k, Frag16& a, Frag16& w0, Frag16& w1, Frag16& w2, Frag16& w3) {
    a.u[0]  = *(const uint4*)(pA  + k);
    a.u[1]  = *(const uint4*)(pA  + k + 8);
    w0.u[0] = *(const uint4*)(pW0 + k);
    w0.u[1] = *(const uint4*)(pW0 + k + 8);
    w1.u[0] = *(const uint4*)(pW1 + k);
    w1.u[1] = *(const uint4*)(pW1 + k + 8);
    w2.u[0] = *(const uint4*)(pW2 + k);
    w2.u[1] = *(const uint4*)(pW2 + k + 8);
    w3.u[0] = *(const uint4*)(pW3 + k);
    w3.u[1] = *(const uint4*)(pW3 + k + 8);
    __builtin_prefetch(pA  + k + 128, 0, 3);   // near-cache prefetch, 4 tiles ahead
    __builtin_prefetch(pW0 + k + 128, 0, 3);
    __builtin_prefetch(pW2 + k + 128, 0, 3);
  };
  auto mm4 = [&](const Frag16& a, const Frag16& w0, const Frag16& w1,
                 const Frag16& w2, const Frag16& w3) {
    acc0 = __builtin_amdgcn_wmma_f32_16x16x32_bf16(false, a.v, false, w0.v, (short)0, acc0, false, false);
    acc1 = __builtin_amdgcn_wmma_f32_16x16x32_bf16(false, a.v, false, w1.v, (short)0, acc1, false, false);
    acc2 = __builtin_amdgcn_wmma_f32_16x16x32_bf16(false, a.v, false, w2.v, (short)0, acc2, false, false);
    acc3 = __builtin_amdgcn_wmma_f32_16x16x32_bf16(false, a.v, false, w3.v, (short)0, acc3, false, false);
  };

  Frag16 aX, w0X, w1X, w2X, w3X;
  Frag16 aY, w0Y, w1Y, w2Y, w3Y;
  const int ns = K >> 5;            // 24 or 48 (always even)
  lda(0, aX, w0X, w1X, w2X, w3X);
  int i = 0;
  for (; i + 2 < ns; i += 2) {      // branch-free steady state
    lda((i + 1) << 5, aY, w0Y, w1Y, w2Y, w3Y);
    mm4(aX, w0X, w1X, w2X, w3X);
    lda((i + 2) << 5, aX, w0X, w1X, w2X, w3X);
    mm4(aY, w0Y, w1Y, w2Y, w3Y);
  }
  lda((ns - 1) << 5, aY, w0Y, w1Y, w2Y, w3Y);   // epilogue pair
  mm4(aX, w0X, w1X, w2X, w3X);
  mm4(aY, w0Y, w1Y, w2Y, w3Y);

  // Epilogue: C/D 16x16 f32 layout: VGPR r -> row m0+r(+8 for hi), col = lo.
  v8f accs[4] = {acc0, acc1, acc2, acc3};
#pragma unroll
  for (int nt = 0; nt < 4; ++nt) {
    const int ncol = n0 + nt * 16 + lo;
    const float bias = b0[ncol] + b1[ncol];
#pragma unroll
    for (int r = 0; r < 8; ++r) {
      const int row = m0 + r + hi * 8;
      C[(size_t)row * G4_ + ncol] = accs[nt][r] + bias;
    }
  }
}

// ---------------------------------------------------------------------------
// Persistent bidirectional LSTM recurrence.
// grid = (24 slices, 2 directions), block = 256 (8 waves).
// Each WG owns 32 hidden units -> 128 gate rows of W_hh kept in LDS (196 KB).
// Per step: stage h (16x768 bf16) to LDS, 8 waves x 24 pipelined bf16 WMMAs,
// gate activations, write h slice, grid barrier.
// ---------------------------------------------------------------------------
#define REC_SMEM (128*768*2 + 16*768*2 + 16*128*4 + 16*32*4)  // 231424 B <= 320KB

__global__ void __launch_bounds__(256)
lstm_rec_kernel(const float* __restrict__ xg_f, const float* __restrict__ xg_r,
                const bf16* __restrict__ whh_f, const bf16* __restrict__ whh_r,
                float* __restrict__ hout, bf16* __restrict__ hout_bf,
                bf16* __restrict__ hstate, int* __restrict__ syncp)
{
  extern __shared__ char smem[];
  bf16*  Wt = (bf16*)smem;                                     // [128][768]
  bf16*  hC = (bf16*)(smem + 128*768*2);                       // [16][768]
  float* gb = (float*)(smem + 128*768*2 + 16*768*2);           // [16][128]
  float* cs = (float*)(smem + 128*768*2 + 16*768*2 + 16*128*4);// [16][32]

  const int dir = blockIdx.y;
  const float* xg  = dir ? xg_r : xg_f;
  const bf16*  whh = dir ? whh_r : whh_f;
  bf16* hst = hstate + (size_t)dir * B_ * H_;
  int* cnt = syncp + dir * 2;
  int* gen = cnt + 1;
  const int j0   = blockIdx.x * 32;
  const int hoff = dir ? H_ : 0;

  const int tid  = threadIdx.x;
  const int wave = tid >> 5, lane = tid & 31, hi = lane >> 4, lo = lane & 15;

  // Preload this slice's 128 rows of W_hh into LDS (rows g*H + j0 + jj).
  for (int idx = tid; idx < (128 * 768) / 8; idx += 256) {
    const int e = idx * 8, r = e / 768, k = e % 768;
    const int g = r >> 5, jj = r & 31;
    *(uint4*)(Wt + r * 768 + k) =
        *(const uint4*)(whh + (size_t)(g * H_ + j0 + jj) * H_ + k);
  }
  // Zero h-state slice (global) and c-state (LDS).
  for (int idx = tid; idx < 512; idx += 256) {
    const int b = idx >> 5, jj = idx & 31;
    hst[b * H_ + j0 + jj] = (bf16)0.0f;
    cs[idx] = 0.0f;
  }
  __syncthreads();
  grid_sync(cnt, gen, 24);

  const int r = wave * 16 + lo;     // gate column within this slice (0..127)
  const bf16* paB = hC + (size_t)lo * H_ + hi * 16;
  const bf16* pwB = Wt + (size_t)r  * H_ + hi * 16;

  for (int s = 0; s < S_; ++s) {
    const int t = dir ? (S_ - 1 - s) : s;

    // Stage full h (all slices, written last step) global -> LDS.
    for (int idx = tid; idx < (B_ * H_) / 8; idx += 256)
      *(uint4*)(hC + idx * 8) = *(const uint4*)(hst + idx * 8);
    __syncthreads();

    auto ldr = [&](int k, Frag16& a, Frag16& w) {
      a.u[0] = *(const uint4*)(paB + k);
      a.u[1] = *(const uint4*)(paB + k + 8);
      w.u[0] = *(const uint4*)(pwB + k);
      w.u[1] = *(const uint4*)(pwB + k + 8);
    };

    // h(16x768) @ Whh_slice^T -> one 16x16 tile per wave.
    // Two-deep ping-pong pipeline over the 24 K-steps (compile-time count).
    v8f acc = {0.f,0.f,0.f,0.f,0.f,0.f,0.f,0.f};
    Frag16 aX, wX, aY, wY;
    ldr(0, aX, wX);
#pragma unroll
    for (int i = 0; i + 2 < 24; i += 2) {
      ldr((i + 1) << 5, aY, wY);
      acc = __builtin_amdgcn_wmma_f32_16x16x32_bf16(false, aX.v, false, wX.v, (short)0, acc, false, false);
      ldr((i + 2) << 5, aX, wX);
      acc = __builtin_amdgcn_wmma_f32_16x16x32_bf16(false, aY.v, false, wY.v, (short)0, acc, false, false);
    }
    ldr(23 << 5, aY, wY);
    acc = __builtin_amdgcn_wmma_f32_16x16x32_bf16(false, aX.v, false, wX.v, (short)0, acc, false, false);
    acc = __builtin_amdgcn_wmma_f32_16x16x32_bf16(false, aY.v, false, wY.v, (short)0, acc, false, false);

    // gates = acc + precomputed xg (biases already folded in) -> LDS.
    {
      const int col = r;                       // 0..127 within slice
      const int g = col >> 5, jj = col & 31;
      const size_t gcol = (size_t)g * H_ + j0 + jj;
#pragma unroll
      for (int e = 0; e < 8; ++e) {
        const int b = e + hi * 8;
        gb[b * 128 + col] = acc[e] + xg[((size_t)b * S_ + t) * G4_ + gcol];
      }
    }
    __syncthreads();

    // Activations + state update for (b, jj) pairs.
    for (int idx = tid; idx < 512; idx += 256) {
      const int b = idx >> 5, jj = idx & 31;
      const float iv = gb[b * 128 +       jj];
      const float fv = gb[b * 128 +  32 + jj];
      const float gv = gb[b * 128 +  64 + jj];
      const float ov = gb[b * 128 +  96 + jj];
      const float c  = sigmoidf_(fv) * cs[idx] + sigmoidf_(iv) * tanhf(gv);
      const float h  = sigmoidf_(ov) * tanhf(c);
      cs[idx] = c;
      hst[b * H_ + j0 + jj] = (bf16)h;
      const size_t o = ((size_t)b * S_ + t) * H2_ + hoff + j0 + jj;
      hout[o]    = h;
      hout_bf[o] = (bf16)h;
    }
    grid_sync(cnt, gen, 24);
  }
}

// ---------------------------------------------------------------------------
// logits[b,t,l] = tanh(h2[b,t,:]) . clf_w[l,:] + clf_b[l]   (N=25, K=1536)
// Block 256 = 8 waves, wave handles one (b,t) row. Lane-split dot + shfl reduce.
// ---------------------------------------------------------------------------
__global__ void __launch_bounds__(256)
clf_kernel(const float* __restrict__ h2, const float* __restrict__ Wc,
           const float* __restrict__ bc, float* __restrict__ logits)
{
  const int wave = threadIdx.x >> 5, lane = threadIdx.x & 31;
  const int row  = blockIdx.x * 8 + wave;
  const float* hr = h2 + (size_t)row * H2_;
  float th[H2_ / 32];
#pragma unroll
  for (int q = 0; q < H2_ / 32; ++q) th[q] = tanhf(hr[q * 32 + lane]);
  for (int l = 0; l < L_; ++l) {
    const float* wl = Wc + (size_t)l * H2_;
    float sacc = 0.f;
#pragma unroll
    for (int q = 0; q < H2_ / 32; ++q) sacc += th[q] * wl[q * 32 + lane];
#pragma unroll
    for (int off = 16; off > 0; off >>= 1) sacc += __shfl_xor(sacc, off, 32);
    if (lane == 0) logits[(size_t)row * L_ + l] = sacc + bc[l];
  }
}

// ---------------------------------------------------------------------------
// CRF negative log-likelihood: single 512-thread block.
// Thread (b = tid>>5, j = tid&31) owns alpha[b][j]; LDS double buffer.
// ---------------------------------------------------------------------------
__global__ void __launch_bounds__(512)
crf_kernel(const float* __restrict__ logits, const int* __restrict__ targets,
           const float* __restrict__ trans, const float* __restrict__ startv,
           const float* __restrict__ endv, float* __restrict__ out)
{
  __shared__ float al[2][B_ * 32];
  __shared__ float tr[L_ * L_];
  __shared__ float num[B_], den[B_];
  __shared__ int   len[B_];

  const int tid = threadIdx.x;
  const int b = tid >> 5, j = tid & 31;

  for (int i = tid; i < L_ * L_; i += 512) tr[i] = trans[i];
  __syncthreads();

  // Per-batch sequential numerator + length (16 threads).
  if (tid < B_) {
    const int bb = tid;
    int Lb = 0;
    for (int t = 0; t < S_; ++t) Lb += (targets[bb * S_ + t] != -1);
    len[bb] = Lb;
    int tg0 = targets[bb * S_ + 0]; if (tg0 < 0) tg0 = 0;
    float nu = startv[tg0] + logits[((size_t)bb * S_ + 0) * L_ + tg0];
    int prev = tg0;
    for (int t = 1; t < S_; ++t) {
      const int tg = targets[bb * S_ + t];
      const int m  = (tg != -1);
      const int tc = m ? tg : 0;
      const float sc = tr[prev * L_ + tc] + logits[((size_t)bb * S_ + t) * L_ + tc];
      nu += m ? sc : 0.f;
      prev = tc;
    }
    const int lt = targets[bb * S_ + (len[bb] - 1)];
    nu += endv[lt < 0 ? 0 : lt];
    num[bb] = nu;
  }

  // alpha0 = start + emissions[0]
  if (j < L_) al[0][b * 32 + j] = startv[j] + logits[((size_t)b * S_ + 0) * L_ + j];
  __syncthreads();

  int cur = 0;
  for (int t = 1; t < S_; ++t) {
    if (j < L_) {
      const int m = (targets[b * S_ + t] != -1);
      float mx = -1e30f;
#pragma unroll
      for (int i = 0; i < L_; ++i)
        mx = fmaxf(mx, al[cur][b * 32 + i] + tr[i * L_ + j]);
      float ss = 0.f;
#pragma unroll
      for (int i = 0; i < L_; ++i)
        ss += __expf(al[cur][b * 32 + i] + tr[i * L_ + j] - mx);
      const float nxt = mx + __logf(ss) + logits[((size_t)b * S_ + t) * L_ + j];
      al[cur ^ 1][b * 32 + j] = m ? nxt : al[cur][b * 32 + j];
    }
    __syncthreads();
    cur ^= 1;
  }

  if (tid < B_) {
    const int bb = tid;
    float mx = -1e30f;
    for (int jj = 0; jj < L_; ++jj) mx = fmaxf(mx, al[cur][bb * 32 + jj] + endv[jj]);
    float ss = 0.f;
    for (int jj = 0; jj < L_; ++jj) ss += __expf(al[cur][bb * 32 + jj] + endv[jj] - mx);
    den[bb] = mx + __logf(ss);
  }
  __syncthreads();
  if (tid == 0) {
    float llh = 0.f, msum = 0.f;
    for (int bb = 0; bb < B_; ++bb) { llh += num[bb] - den[bb]; msum += (float)len[bb]; }
    out[0] = -(llh / msum);
  }
}

// ---------------------------------------------------------------------------
// Host driver
// ---------------------------------------------------------------------------
extern "C" void kernel_launch(void* const* d_in, const int* in_sizes, int n_in,
                              void* d_out, int out_size, void* d_ws, size_t ws_size,
                              hipStream_t stream)
{
  (void)in_sizes; (void)n_in; (void)out_size; (void)ws_size;

  const float* hidden    = (const float*)d_in[0];
  const int*   targets   = (const int*)  d_in[1];
  const float* w_ih_l0   = (const float*)d_in[2];
  const float* w_hh_l0   = (const float*)d_in[3];
  const float* b_ih_l0   = (const float*)d_in[4];
  const float* b_hh_l0   = (const float*)d_in[5];
  const float* w_ih_l0r  = (const float*)d_in[6];
  const float* w_hh_l0r  = (const float*)d_in[7];
  const float* b_ih_l0r  = (const float*)d_in[8];
  const float* b_hh_l0r  = (const float*)d_in[9];
  const float* w_ih_l1   = (const float*)d_in[10];
  const float* w_hh_l1   = (const float*)d_in[11];
  const float* b_ih_l1   = (const float*)d_in[12];
  const float* b_hh_l1   = (const float*)d_in[13];
  const float* w_ih_l1r  = (const float*)d_in[14];
  const float* w_hh_l1r  = (const float*)d_in[15];
  const float* b_ih_l1r  = (const float*)d_in[16];
  const float* b_hh_l1r  = (const float*)d_in[17];
  const float* clf_w     = (const float*)d_in[18];
  const float* clf_b     = (const float*)d_in[19];
  const float* crf_trans = (const float*)d_in[20];
  const float* crf_start = (const float*)d_in[21];
  const float* crf_end   = (const float*)d_in[22];

  char* ws = (char*)d_ws;
  size_t off = 0;
  auto take = [&](size_t bytes) -> char* {
    char* p = ws + off;
    off = (off + bytes + 255) & ~(size_t)255;
    return p;
  };

  int*   syncp  = (int*)  take(256);
  bf16*  x_bf   = (bf16*) take((size_t)B_ * S_ * H_ * 2);
  bf16*  wih0   = (bf16*) take((size_t)G4_ * H_  * 2);
  bf16*  wih0r  = (bf16*) take((size_t)G4_ * H_  * 2);
  bf16*  whh0   = (bf16*) take((size_t)G4_ * H_  * 2);
  bf16*  whh0r  = (bf16*) take((size_t)G4_ * H_  * 2);
  bf16*  whh1   = (bf16*) take((size_t)G4_ * H_  * 2);
  bf16*  whh1r  = (bf16*) take((size_t)G4_ * H_  * 2);
  bf16*  wih1   = (bf16*) take((size_t)G4_ * H2_ * 2);
  bf16*  wih1r  = (bf16*) take((size_t)G4_ * H2_ * 2);
  float* xg_a   = (float*)take((size_t)B_ * S_ * G4_ * 4);
  float* xg_b   = (float*)take((size_t)B_ * S_ * G4_ * 4);
  float* h1f    = (float*)take((size_t)B_ * S_ * H2_ * 4);
  bf16*  h1b    = (bf16*) take((size_t)B_ * S_ * H2_ * 2);
  float* h2f    = (float*)take((size_t)B_ * S_ * H2_ * 4);
  bf16*  h2b    = (bf16*) take((size_t)B_ * S_ * H2_ * 2);
  bf16*  hstate = (bf16*) take((size_t)2 * B_ * H_ * 2);
  float* logits = (float*)take((size_t)B_ * S_ * L_ * 4);

  (void)hipFuncSetAttribute((const void*)lstm_rec_kernel,
                            hipFuncAttributeMaxDynamicSharedMemorySize, REC_SMEM);

  auto cast = [&](const float* s, bf16* d, size_t n) {
    unsigned blocks = (unsigned)((n + 2047) / 2048);
    cast_bf16_kernel<<<blocks, 256, 0, stream>>>(s, d, n);
  };
  cast(hidden,    x_bf,  (size_t)B_ * S_ * H_);
  cast(w_ih_l0,   wih0,  (size_t)G4_ * H_);
  cast(w_ih_l0r,  wih0r, (size_t)G4_ * H_);
  cast(w_hh_l0,   whh0,  (size_t)G4_ * H_);
  cast(w_hh_l0r,  whh0r, (size_t)G4_ * H_);
  cast(w_hh_l1,   whh1,  (size_t)G4_ * H_);
  cast(w_hh_l1r,  whh1r, (size_t)G4_ * H_);
  cast(w_ih_l1,   wih1,  (size_t)G4_ * H2_);
  cast(w_ih_l1r,  wih1r, (size_t)G4_ * H2_);

  const dim3 ggrid(G4_ / 512, (B_ * S_) / 16);   // (6, 512)

  // Layer 0: input projections, then persistent bidirectional recurrence.
  gemm_xg_kernel<<<ggrid, 256, 0, stream>>>(x_bf, wih0,  b_ih_l0,  b_hh_l0,  xg_a, H_);
  gemm_xg_kernel<<<ggrid, 256, 0, stream>>>(x_bf, wih0r, b_ih_l0r, b_hh_l0r, xg_b, H_);
  (void)hipMemsetAsync(syncp, 0, 256, stream);
  lstm_rec_kernel<<<dim3(24, 2), 256, REC_SMEM, stream>>>(
      xg_a, xg_b, whh0, whh0r, h1f, h1b, hstate, syncp);

  // Layer 1 (input = concat h1, K = 1536).
  gemm_xg_kernel<<<ggrid, 256, 0, stream>>>(h1b, wih1,  b_ih_l1,  b_hh_l1,  xg_a, H2_);
  gemm_xg_kernel<<<ggrid, 256, 0, stream>>>(h1b, wih1r, b_ih_l1r, b_hh_l1r, xg_b, H2_);
  (void)hipMemsetAsync(syncp, 0, 256, stream);
  lstm_rec_kernel<<<dim3(24, 2), 256, REC_SMEM, stream>>>(
      xg_a, xg_b, whh1, whh1r, h2f, h2b, hstate, syncp);

  // Classifier + CRF NLL.
  clf_kernel<<<dim3((B_ * S_) / 8), 256, 0, stream>>>(h2f, clf_w, clf_b, logits);
  crf_kernel<<<1, 512, 0, stream>>>(logits, targets, crf_trans, crf_start, crf_end,
                                    (float*)d_out);
}
